// PhysicsInformedESN_41145786695770
// MI455X (gfx1250) — matrix operand order repaired and checked by reading, not compiled, not run on hardware
//
#include <hip/hip_runtime.h>
#include <hip/hip_bf16.h>
#include <math.h>

#define GRID   64
#define BLK    256
#define Nres   2048
#define DIN    32
#define DOUT   16
#define Hdim   128
#define Bsz    64
#define Tlen   256

typedef unsigned short u16;
typedef __attribute__((ext_vector_type(16))) __bf16 v16bf;
typedef __attribute__((ext_vector_type(8)))  __bf16 v8bf;
typedef __attribute__((ext_vector_type(8)))  float  v8f;

__device__ __forceinline__ u16 f2bf(float f) {
  unsigned u = __float_as_uint(f);
  u += 0x7FFFu + ((u >> 16) & 1u);   // round-to-nearest-even
  return (u16)(u >> 16);
}

__device__ __forceinline__ float gelu_exact(float x) {
  return 0.5f * x * (1.0f + erff(x * 0.70710678118654752440f));
}

// A fragment: 16x32 bf16, row-major source (CDNA5 ISA 7.12.2 layout).
__device__ __forceinline__ v16bf load_a_frag(const u16* __restrict__ base, int ld,
                                             int m0, int k0, int lane) {
  int r  = lane & 15;
  int kb = (lane & 16) ? 8 : 0;
  const u16* p = base + (size_t)(m0 + r) * ld + (k0 + kb);
  v8bf lo = *(const v8bf*)p;
  v8bf hi = *(const v8bf*)(p + 16);
  return __builtin_shufflevector(lo, hi, 0,1,2,3,4,5,6,7,8,9,10,11,12,13,14,15);
}

// B fragment: 32x16 bf16 = transposed weight; source row-major [Nrows, K].
__device__ __forceinline__ v16bf load_b_frag(const u16* __restrict__ base, int ld,
                                             int n0, int k0, int lane) {
  int c  = lane & 15;
  int kh = (lane & 16) ? 8 : 0;
  const u16* p = base + (size_t)(n0 + c) * ld + (k0 + kh);
  v8bf lo = *(const v8bf*)p;
  v8bf hi = *(const v8bf*)(p + 16);
  return __builtin_shufflevector(lo, hi, 0,1,2,3,4,5,6,7,8,9,10,11,12,13,14,15);
}

__device__ __forceinline__ v8f wmma_loop(const u16* __restrict__ Am, int lda,
                                         const u16* __restrict__ Bm, int ldb,
                                         int m0, int n0, int K, int lane, v8f acc) {
#pragma unroll 4
  for (int k0 = 0; k0 < K; k0 += 32) {
    v16bf a = load_a_frag(Am, lda, m0, k0, lane);
    v16bf b = load_b_frag(Bm, ldb, n0, k0, lane);
    acc = __builtin_amdgcn_wmma_f32_16x16x32_bf16(false, a, false, b,
                                                  (short)0, acc, false, false);
  }
  return acc;
}

__device__ __forceinline__ void grid_barrier(unsigned* cnt, unsigned target) {
  __threadfence();
  __syncthreads();
  if (threadIdx.x == 0) {
    atomicAdd(cnt, 1u);
    while (__hip_atomic_load(cnt, __ATOMIC_ACQUIRE, __HIP_MEMORY_SCOPE_AGENT) < target)
      __builtin_amdgcn_s_sleep(4);
  }
  __syncthreads();
}

__global__ void cvt_f32_bf16(const float* __restrict__ s, u16* __restrict__ d, int n) {
  int i = blockIdx.x * blockDim.x + threadIdx.x;
  if (i < n) d[i] = f2bf(s[i]);
}

// d[c*rows + r] = s[r*cols + c]  (transpose + convert, one-time)
__global__ void transpose_cvt(const float* __restrict__ s, u16* __restrict__ d,
                              int rows, int cols) {
  int i = blockIdx.x * blockDim.x + threadIdx.x;
  if (i < rows * cols) {
    int r = i / cols, c = i % cols;
    d[(size_t)c * rows + r] = f2bf(s[i]);
  }
}

// b2[row] = dot(Wproj[row,:], bv) + bproj[row]   (fp32, one-time)
__global__ __launch_bounds__(256) void fuse_bias(const float* __restrict__ Wproj,
                                                 const float* __restrict__ bv,
                                                 const float* __restrict__ bproj,
                                                 float* __restrict__ b2) {
  __shared__ float red[256];
  int row = blockIdx.x;
  const float* wr = Wproj + (size_t)row * Nres;
  float s = 0.0f;
  for (int k = threadIdx.x; k < Nres; k += 256) s = fmaf(wr[k], bv[k], s);
  red[threadIdx.x] = s;
  __syncthreads();
  for (int off = 128; off > 0; off >>= 1) {
    if (threadIdx.x < off) red[threadIdx.x] += red[threadIdx.x + off];
    __syncthreads();
  }
  if (threadIdx.x == 0) b2[row] = red[0] + bproj[row];
}

// W2 = Wproj @ Wv  (bf16 WMMA, one-time). Aw = Wproj bf16 row-major [N,N];
// Bw = Wv^T bf16 row-major [N(out j), K] so B[k,n] = Wv[k][n]. 16384 tiles.
__global__ __launch_bounds__(BLK) void fuse_w2(const u16* __restrict__ Aw,
                                               const u16* __restrict__ Bw,
                                               u16* __restrict__ W2) {
  int lane = threadIdx.x & 31;
  int wid  = blockIdx.x * (BLK / 32) + (threadIdx.x >> 5);  // 0..16383
  int m0 = (wid >> 7) << 4;
  int n0 = (wid & 127) << 4;
  v8f acc = {0,0,0,0,0,0,0,0};
  acc = wmma_loop(Aw, Nres, Bw, Nres, m0, n0, Nres, lane, acc);
  int c  = n0 + (lane & 15);
  int mb = m0 + ((lane & 16) ? 8 : 0);
#pragma unroll
  for (int r = 0; r < 8; ++r)
    W2[(size_t)(mb + r) * Nres + c] = f2bf(acc[r]);
}

struct EsnArgs {
  const u16 *Xbf, *Win, *Wres, *W2, *We1, *We2, *We3;
  const float *b2, *be1, *be2, *be3, *ln_g, *ln_b;
  const float *w_mu, *w_rho, *b_mu, *b_rho, *w_eps, *b_eps;
  float *S; u16 *Sbf; float *R; float *Satt; u16 *Sattbf;
  float *H1; u16 *H1n; u16 *H2;
  unsigned *cnt;
  float *out;
};

__global__ __launch_bounds__(BLK) void esn_persistent(EsnArgs A) {
  const int blk  = blockIdx.x;
  const int lane = threadIdx.x & 31;
  const int wave = blk * (BLK / 32) + (threadIdx.x >> 5);   // 0..511
  unsigned epoch = 0;

  for (int t = 0; t < Tlen; ++t) {
    // ---- Stage A: R = x_t@Win^T + S@Wres^T ; Satt = S@W2^T + b2 ----
#pragma unroll
    for (int tt = 0; tt < 2; ++tt) {
      int tid = wave * 2 + tt;          // 0..1023 logical tiles
      int idx = tid & 511;
      int m0 = (idx >> 7) << 4;
      int n0 = (idx & 127) << 4;
      v8f acc = {0,0,0,0,0,0,0,0};
      int c  = n0 + (lane & 15);
      int mb = m0 + ((lane & 16) ? 8 : 0);
      if (tid < 512) {
        acc = wmma_loop(A.Xbf + (size_t)t * DIN, Tlen * DIN, A.Win, DIN,
                        m0, n0, DIN, lane, acc);
        acc = wmma_loop(A.Sbf, Nres, A.Wres, Nres, m0, n0, Nres, lane, acc);
#pragma unroll
        for (int r = 0; r < 8; ++r)
          A.R[(size_t)(mb + r) * Nres + c] = acc[r];
      } else {
        acc = wmma_loop(A.Sbf, Nres, A.W2, Nres, m0, n0, Nres, lane, acc);
        float bb = A.b2[c];
#pragma unroll
        for (int r = 0; r < 8; ++r) {
          float v = acc[r] + bb;
          size_t o = (size_t)(mb + r) * Nres + c;
          A.Satt[o]   = v;
          A.Sattbf[o] = f2bf(v);
        }
      }
    }
    grid_barrier(A.cnt, ++epoch * GRID);

    // ---- Stage B: H1 = Satt@We1^T + be1  (4x8 = 32 tiles) ----
    if (wave < 32) {
      int m0 = (wave >> 3) << 4;
      int n0 = (wave & 7) << 4;
      v8f acc = {0,0,0,0,0,0,0,0};
      acc = wmma_loop(A.Sattbf, Nres, A.We1, Nres, m0, n0, Nres, lane, acc);
      int c  = n0 + (lane & 15);
      int mb = m0 + ((lane & 16) ? 8 : 0);
      float bb = A.be1[c];
#pragma unroll
      for (int r = 0; r < 8; ++r)
        A.H1[(size_t)(mb + r) * Hdim + c] = acc[r] + bb;
    }
    grid_barrier(A.cnt, ++epoch * GRID);

    // ---- Stage C (block 0): LayerNorm+GELU then H2 = gelu(H1n@We2^T+be2) ----
    if (blk == 0) {
      int w = threadIdx.x >> 5;
#pragma unroll
      for (int rr = 0; rr < 8; ++rr) {
        int row = w * 8 + rr;
        const float* hr = A.H1 + (size_t)row * Hdim;
        float v0 = hr[lane], v1 = hr[lane + 32], v2 = hr[lane + 64], v3 = hr[lane + 96];
        float s  = v0 + v1 + v2 + v3;
        float sq = v0*v0 + v1*v1 + v2*v2 + v3*v3;
#pragma unroll
        for (int m = 16; m >= 1; m >>= 1) {
          s  += __shfl_xor(s,  m, 32);
          sq += __shfl_xor(sq, m, 32);
        }
        float mean = s * (1.0f / Hdim);
        float var  = sq * (1.0f / Hdim) - mean * mean;
        float rstd = rsqrtf(var + 1e-5f);
        u16* dr = A.H1n + (size_t)row * Hdim;
        float vv[4] = {v0, v1, v2, v3};
#pragma unroll
        for (int i = 0; i < 4; ++i) {
          int k = lane + i * 32;
          float hn = (vv[i] - mean) * rstd * A.ln_g[k] + A.ln_b[k];
          dr[k] = f2bf(gelu_exact(hn));
        }
      }
      __syncthreads();
#pragma unroll
      for (int q = 0; q < 4; ++q) {
        int idx = w * 4 + q;               // 32 tiles over 8 waves
        int m0 = (idx >> 3) << 4;
        int n0 = (idx & 7) << 4;
        v8f acc = {0,0,0,0,0,0,0,0};
        acc = wmma_loop(A.H1n, Hdim, A.We2, Hdim, m0, n0, Hdim, lane, acc);
        int c  = n0 + (lane & 15);
        int mb = m0 + ((lane & 16) ? 8 : 0);
        float bb = A.be2[c];
#pragma unroll
        for (int r = 0; r < 8; ++r)
          A.H2[(size_t)(mb + r) * Hdim + c] = f2bf(gelu_exact(acc[r] + bb));
      }
    }
    grid_barrier(A.cnt, ++epoch * GRID);

    // ---- Stage D: phys = H2@We3^T + be3 ; S = 0.7*Satt + 0.3*tanh(R+phys) ----
    {
      int m0 = (wave >> 7) << 4;
      int n0 = (wave & 127) << 4;
      v8f acc = {0,0,0,0,0,0,0,0};
      acc = wmma_loop(A.H2, Hdim, A.We3, Hdim, m0, n0, Hdim, lane, acc);
      int c  = n0 + (lane & 15);
      int mb = m0 + ((lane & 16) ? 8 : 0);
      float bb = A.be3[c];
#pragma unroll
      for (int r = 0; r < 8; ++r) {
        size_t o = (size_t)(mb + r) * Nres + c;
        float val = A.R[o] + acc[r] + bb;
        float ns  = 0.7f * A.Satt[o] + 0.3f * tanhf(val);
        A.S[o]   = ns;
        A.Sbf[o] = f2bf(ns);
      }
    }
    grid_barrier(A.cnt, ++epoch * GRID);
  }

  // ---- Final Bayesian readout: [64,16] = S @ weight^T + bias (block 0, fp32) ----
  if (blk == 0) {
    for (int idx = threadIdx.x; idx < Bsz * DOUT; idx += BLK) {
      int b = idx >> 4, o = idx & 15;
      const float* wm = A.w_mu  + (size_t)o * Nres;
      const float* wr = A.w_rho + (size_t)o * Nres;
      const float* we = A.w_eps + (size_t)o * Nres;
      const float* sr = A.S     + (size_t)b * Nres;
      float accv = 0.0f;
      for (int k = 0; k < Nres; ++k)
        accv = fmaf(sr[k], wm[k] + log1pf(expf(wr[k])) * we[k], accv);
      float bias = A.b_mu[o] + log1pf(expf(A.b_rho[o])) * A.b_eps[o];
      A.out[idx] = accv + bias;
    }
  }
}

extern "C" void kernel_launch(void* const* d_in, const int* in_sizes, int n_in,
                              void* d_out, int out_size, void* d_ws, size_t ws_size,
                              hipStream_t stream) {
  const float* x     = (const float*)d_in[0];
  const float* W_in  = (const float*)d_in[1];
  const float* W_res = (const float*)d_in[2];
  const float* Wv    = (const float*)d_in[3];
  const float* bv    = (const float*)d_in[4];
  const float* Wproj = (const float*)d_in[5];
  const float* bproj = (const float*)d_in[6];
  const float* We1   = (const float*)d_in[7];
  const float* be1   = (const float*)d_in[8];
  const float* ln_g  = (const float*)d_in[9];
  const float* ln_b  = (const float*)d_in[10];
  const float* We2   = (const float*)d_in[11];
  const float* be2   = (const float*)d_in[12];
  const float* We3   = (const float*)d_in[13];
  const float* be3   = (const float*)d_in[14];
  const float* w_mu  = (const float*)d_in[15];
  const float* w_rho = (const float*)d_in[16];
  const float* b_mu  = (const float*)d_in[17];
  const float* b_rho = (const float*)d_in[18];
  const float* w_eps = (const float*)d_in[19];
  const float* b_eps = (const float*)d_in[20];

  char* ws = (char*)d_ws;
  size_t off = 0;
  auto take = [&](size_t bytes) -> char* {
    char* p = ws + off;
    off += (bytes + 255) & ~(size_t)255;
    return p;
  };
  unsigned* cnt  = (unsigned*)take(256);
  float* S       = (float*)take((size_t)Bsz * Nres * 4);
  u16*   Sbf     = (u16*)  take((size_t)Bsz * Nres * 2);
  float* R       = (float*)take((size_t)Bsz * Nres * 4);
  float* Satt    = (float*)take((size_t)Bsz * Nres * 4);
  u16*   Sattbf  = (u16*)  take((size_t)Bsz * Nres * 2);
  float* H1      = (float*)take((size_t)Bsz * Hdim * 4);
  u16*   H1n     = (u16*)  take((size_t)Bsz * Hdim * 2);
  u16*   H2      = (u16*)  take((size_t)Bsz * Hdim * 2);
  float* b2      = (float*)take((size_t)Nres * 4);
  u16*   Wresbf  = (u16*)  take((size_t)Nres * Nres * 2);
  u16*   Wvt     = (u16*)  take((size_t)Nres * Nres * 2);   // Wv^T bf16
  u16*   Wprojbf = (u16*)  take((size_t)Nres * Nres * 2);
  u16*   W2bf    = (u16*)  take((size_t)Nres * Nres * 2);   // Wproj@Wv bf16
  u16*   We1bf   = (u16*)  take((size_t)Hdim * Nres * 2);
  u16*   We2bf   = (u16*)  take((size_t)Hdim * Hdim * 2);
  u16*   We3bf   = (u16*)  take((size_t)Nres * Hdim * 2);
  u16*   Winbf   = (u16*)  take((size_t)Nres * DIN * 2);
  u16*   Xbf     = (u16*)  take((size_t)Bsz * Tlen * DIN * 2);

  hipMemsetAsync(cnt, 0, 256, stream);
  hipMemsetAsync(S,   0, (size_t)Bsz * Nres * 4, stream);
  hipMemsetAsync(Sbf, 0, (size_t)Bsz * Nres * 2, stream);

  auto cvt = [&](const float* s, u16* d, int n) {
    cvt_f32_bf16<<<(n + 255) / 256, 256, 0, stream>>>(s, d, n);
  };
  cvt(W_res, Wresbf, Nres * Nres);
  cvt(Wproj, Wprojbf, Nres * Nres);
  cvt(We1,   We1bf,  Hdim * Nres);
  cvt(We2,   We2bf,  Hdim * Hdim);
  cvt(We3,   We3bf,  Nres * Hdim);
  cvt(W_in,  Winbf,  Nres * DIN);
  cvt(x,     Xbf,    Bsz * Tlen * DIN);
  transpose_cvt<<<(Nres * Nres + 255) / 256, 256, 0, stream>>>(Wv, Wvt, Nres, Nres);

  // one-time attention fusion: W2 = Wproj @ Wv, b2 = Wproj@bv + bproj
  fuse_bias<<<Nres, 256, 0, stream>>>(Wproj, bv, bproj, b2);
  fuse_w2<<<(Nres / 16) * (Nres / 16) / (BLK / 32), BLK, 0, stream>>>(Wprojbf, Wvt, W2bf);

  EsnArgs a;
  a.Xbf = Xbf; a.Win = Winbf; a.Wres = Wresbf; a.W2 = W2bf;
  a.We1 = We1bf; a.We2 = We2bf; a.We3 = We3bf;
  a.b2 = b2; a.be1 = be1; a.be2 = be2; a.be3 = be3;
  a.ln_g = ln_g; a.ln_b = ln_b;
  a.w_mu = w_mu; a.w_rho = w_rho; a.b_mu = b_mu; a.b_rho = b_rho;
  a.w_eps = w_eps; a.b_eps = b_eps;
  a.S = S; a.Sbf = Sbf; a.R = R; a.Satt = Satt; a.Sattbf = Sattbf;
  a.H1 = H1; a.H1n = H1n; a.H2 = H2;
  a.cnt = cnt; a.out = (float*)d_out;

  esn_persistent<<<GRID, BLK, 0, stream>>>(a);

  (void)in_sizes; (void)n_in; (void)out_size; (void)ws_size;
}
